// Triangular_Multiplicative_Model_8538394985037
// MI455X (gfx1250) — compile-verified, matrix operand order checked by reading, not run on hardware
//
#include <hip/hip_runtime.h>
#include <hip/hip_bf16.h>

typedef _Float16 v8h  __attribute__((ext_vector_type(8)));
typedef _Float16 v16h __attribute__((ext_vector_type(16)));
typedef float    v8f  __attribute__((ext_vector_type(8)));

#define NSZ 512
#define CZ  128
#define CSM 16
#define NN  (NSZ * NSZ)   // 262144 pixels

// ---- workspace layout (bytes, 256-aligned) ----
#define OFF_WCAT 0u                 // 192*128 f16 = 49152
#define OFF_BCAT 49152u             // 192 f32
#define OFF_AT   50176u             // 16*NN f16 = 8,388,608  (aT[c][i*512+k])
#define OFF_BT   8438784u           // 16*NN f16               (bT[c][j*512+k])
#define OFF_G    16827392u          // NN*128 f16 = 67,108,864
#define OFF_Z    83936256u          // 16*NN f32 = 16,777,216  (zbuf[c][i*512+j])
// total ~100.7 MB

__device__ __forceinline__ float sigm(float v) { return 1.0f / (1.0f + __expf(-v)); }

// Load one WMMA 16-bit operand fragment for this lane from a row-major slab:
// 8 contiguous halves at kbase, 8 more at kbase+16 (per ISA A/B 16-bit layouts).
__device__ __forceinline__ v16h ldfrag(const _Float16* p) {
  v8h lo = *(const v8h*)(p);
  v8h hi = *(const v8h*)(p + 16);
  return __builtin_shufflevector(lo, hi, 0,1,2,3,4,5,6,7,8,9,10,11,12,13,14,15);
}

// Async global->LDS 16-byte copy (CDNA5 GLOBAL_LOAD_ASYNC_TO_LDS_B128, ASYNCcnt).
// lds_dst: generic pointer into __shared__ (low 32 bits = LDS byte address).
__device__ __forceinline__ void async_ld_b128(void* lds_dst, const void* gaddr) {
  unsigned lds = (unsigned)(unsigned long long)(uintptr_t)lds_dst;
  unsigned long long ga = (unsigned long long)(uintptr_t)gaddr;
  asm volatile("global_load_async_to_lds_b128 %0, %1, off"
               :: "v"(lds), "v"(ga) : "memory");
}

__device__ __forceinline__ void wait_asynccnt0() {
  asm volatile("s_wait_asynccnt 0x0" ::: "memory");
}

// ---------------------------------------------------------------------------
// Kernel 0: pack weights into WcatT[192][128] (f16, transposed: k contiguous)
// cols 0-15: Wa1, 16-31: Wa2, 32-47: Wb1, 48-63: Wb2, 64-191: Wg
// ---------------------------------------------------------------------------
__global__ void pack_weights(const float* __restrict__ Wa1, const float* __restrict__ Wa2,
                             const float* __restrict__ Wb1, const float* __restrict__ Wb2,
                             const float* __restrict__ Wg,
                             const float* __restrict__ ba1, const float* __restrict__ ba2,
                             const float* __restrict__ bb1, const float* __restrict__ bb2,
                             const float* __restrict__ bg,
                             _Float16* __restrict__ WcatT, float* __restrict__ bcat) {
  const int col = blockIdx.x;     // 0..191
  const int k   = threadIdx.x;    // 0..127
  float v;
  if      (col <  16) v = Wa1[k * 16 + col];
  else if (col <  32) v = Wa2[k * 16 + (col - 16)];
  else if (col <  48) v = Wb1[k * 16 + (col - 32)];
  else if (col <  64) v = Wb2[k * 16 + (col - 48)];
  else                v = Wg [k * 128 + (col - 64)];
  WcatT[col * 128 + k] = (_Float16)v;
  if (k == 0) {
    float b;
    if      (col <  16) b = ba1[col];
    else if (col <  32) b = ba2[col - 16];
    else if (col <  48) b = bb1[col - 32];
    else if (col <  64) b = bb2[col - 48];
    else                b = bg [col - 64];
    bcat[col] = b;
  }
}

// ---------------------------------------------------------------------------
// Kernel 1: fused projections. One block = 16 rows of x ([NN x 128]).
// GEMM [16x128] @ [128x192] via v_wmma_f32_16x16x32_f16 (4 waves x 3 coltiles).
// Epilogue: a = sig(pa1*pa2) -> aT[c][i*512+k]; b = sig(pb1*pb2) -> bT[c][j*512+k];
//           g = sig(pg) -> gbuf (f16).
// ---------------------------------------------------------------------------
__global__ __launch_bounds__(128) void proj_kernel(
    const float* __restrict__ x, const _Float16* __restrict__ WcatT,
    const float* __restrict__ bcat,
    _Float16* __restrict__ aT, _Float16* __restrict__ bT, _Float16* __restrict__ gbuf) {
  __shared__ _Float16 xs[16][128];   // 4 KB  f16 copy of the 16x128 x tile
  __shared__ float    P[16][192];    // 12 KB raw projection results

  const int  t     = threadIdx.x;
  const long rbase = (long)blockIdx.x * 16;

  // Stage + convert x tile
  for (int e = t; e < 16 * 128; e += 128) {
    const int row = e >> 7, kc = e & 127;
    xs[row][kc] = (_Float16)x[(rbase + row) * CZ + kc];
  }
  __syncthreads();

  const int lane = t & 31, w = t >> 5;
  const int half = lane >> 4, n = lane & 15;
  const _Float16* arow = &xs[n][0];

  for (int q = 0; q < 3; ++q) {
    const int ct = w * 3 + q;                       // col tile 0..11
    const _Float16* brow = WcatT + (ct * 16 + n) * 128;
    v8f acc = {};
    for (int kk = 0; kk < 4; ++kk) {
      const int kbase = kk * 32 + half * 8;
      v16h af = ldfrag(arow + kbase);
      v16h bf = ldfrag(brow + kbase);
      acc = __builtin_amdgcn_wmma_f32_16x16x32_f16(false, af, false, bf,
                                                   (short)0, acc, false, false);
    }
    for (int v = 0; v < 8; ++v)                     // C/D layout: M = v + 8*half, N = n
      P[v + 8 * half][ct * 16 + n] = acc[v];
  }
  __syncthreads();

  // a/b epilogue: sigmoid of the two-projection product, scattered channel-major
  for (int idx = t; idx < 256; idx += 128) {
    const int  c = idx >> 4, row = idx & 15;
    const long rg = rbase + row;                    // rg = first*512 + second
    const float pa = sigm((P[row][c]      + bcat[c])      * (P[row][16 + c] + bcat[16 + c]));
    const float pb = sigm((P[row][32 + c] + bcat[32 + c]) * (P[row][48 + c] + bcat[48 + c]));
    aT[(long)c * NN + rg] = (_Float16)pa;
    const long kq = rg >> 9, j = rg & 511;          // b indexed as [k, j, c]
    bT[(long)c * NN + j * 512 + kq] = (_Float16)pb; // store B transposed for the GEMM
  }
  // g epilogue (coalesced, f16)
  for (int row = 0; row < 16; ++row)
    gbuf[(rbase + row) * CZ + t] = (_Float16)sigm(P[row][64 + t] + bcat[64 + t]);
}

// ---------------------------------------------------------------------------
// Kernel 2: 16 channel-wise 512x512x512 GEMMs: z[c] = A[c] * B[c]^T
// Block = 128x128 tile of one channel, 8 waves x (2x4) 16x16 fragments.
// K chunks staged via async global->LDS copies (ASYNCcnt) with double
// buffering so the copy of chunk k+1 overlaps the WMMAs of chunk k.
// ---------------------------------------------------------------------------
#define LSTR 40   // padded LDS row stride in halves (80 B, 16B-aligned)

__device__ __forceinline__ void stage_chunk(const _Float16* Ap, const _Float16* Bp,
                                            _Float16* AsB, _Float16* BsB,
                                            int t, int kk) {
  for (int rep = 0; rep < 2; ++rep) {
    const int u = t + rep * 256;         // 512 16-byte segments per operand
    const int row = u >> 2, seg = u & 3;
    async_ld_b128(AsB + row * LSTR + seg * 8, Ap + row * 512 + kk * 32 + seg * 8);
    async_ld_b128(BsB + row * LSTR + seg * 8, Bp + row * 512 + kk * 32 + seg * 8);
  }
}

__global__ __launch_bounds__(256) void tri_gemm(
    const _Float16* __restrict__ aT, const _Float16* __restrict__ bT,
    float* __restrict__ zbuf) {
  __shared__ _Float16 As[2][128 * LSTR];  // 2 x 10 KB ping-pong
  __shared__ _Float16 Bs[2][128 * LSTR];  // 2 x 10 KB ping-pong

  const int t  = threadIdx.x;
  const int jt = blockIdx.x, it = blockIdx.y, c = blockIdx.z;
  const _Float16* Ap = aT + (long)c * NN + (long)it * 128 * 512;
  const _Float16* Bp = bT + (long)c * NN + (long)jt * 128 * 512;

  const int lane = t & 31, w = t >> 5;
  const int half = lane >> 4, n = lane & 15;
  const int mt0 = (w & 3) * 2;   // this wave: row frags mt0, mt0+1
  const int nt0 = (w >> 2) * 4;  // this wave: col frags nt0..nt0+3

  v8f acc[2][4] = {};

  stage_chunk(Ap, Bp, As[0], Bs[0], t, 0);
  wait_asynccnt0();
  __syncthreads();

  for (int kk = 0; kk < 16; ++kk) {
    const int cur = kk & 1;
    if (kk < 15) stage_chunk(Ap, Bp, As[cur ^ 1], Bs[cur ^ 1], t, kk + 1);

    // Compute on the current buffer. The ds_load fragments are consumed by the
    // WMMAs (DScnt waited) before this wave reaches the end-of-iteration
    // barrier, so the ping-pong overwrite is race-free.
    const int kb = half * 8;
    v16h afrag[2], bfrag[4];
    for (int m = 0; m < 2; ++m)
      afrag[m] = ldfrag(&As[cur][((mt0 + m) * 16 + n) * LSTR + kb]);
    for (int q = 0; q < 4; ++q)
      bfrag[q] = ldfrag(&Bs[cur][((nt0 + q) * 16 + n) * LSTR + kb]);
    for (int m = 0; m < 2; ++m)
      for (int q = 0; q < 4; ++q)
        acc[m][q] = __builtin_amdgcn_wmma_f32_16x16x32_f16(false, afrag[m], false, bfrag[q],
                                                           (short)0, acc[m][q], false, false);

    if (kk < 15) {
      wait_asynccnt0();   // next buffer fully in LDS (this wave's copies)
      __syncthreads();    // ... and everyone else's
    }
  }

  for (int m = 0; m < 2; ++m)
    for (int q = 0; q < 4; ++q)
      for (int v = 0; v < 8; ++v) {
        const int i = it * 128 + (mt0 + m) * 16 + v + 8 * half;
        const int j = jt * 128 + (nt0 + q) * 16 + n;
        zbuf[(long)c * NN + (long)i * 512 + j] = acc[m][q][v];
      }
}

// ---------------------------------------------------------------------------
// Kernel 3: LayerNorm over C=16 + thin GEMM @Wz + bias + sigmoid-gate.
// One block (128 threads) per pixel; thread t owns output channel t.
// ---------------------------------------------------------------------------
__global__ __launch_bounds__(128) void ln_out_kernel(
    const float* __restrict__ zbuf, const _Float16* __restrict__ gbuf,
    const float* __restrict__ ln_g, const float* __restrict__ ln_b,
    const float* __restrict__ Wz, const float* __restrict__ bz,
    float* __restrict__ out) {
  __shared__ float zs[16];
  const long r = blockIdx.x;
  const int  t = threadIdx.x;
  if (t < 16) zs[t] = zbuf[(long)t * NN + r];
  __syncthreads();

  float zv[16], m = 0.f;
  for (int c = 0; c < 16; ++c) { zv[c] = zs[c]; m += zv[c]; }
  m *= (1.0f / 16.0f);
  float var = 0.f;
  for (int c = 0; c < 16; ++c) { const float d = zv[c] - m; var += d * d; }
  var *= (1.0f / 16.0f);
  const float rstd = __frsqrt_rn(var + 1e-5f);

  float acc = bz[t];
  for (int c = 0; c < 16; ++c) {
    const float zn = (zv[c] - m) * rstd * ln_g[c] + ln_b[c];
    acc += zn * Wz[c * CZ + t];
  }
  out[r * CZ + t] = (float)gbuf[r * CZ + t] * acc;
}

// ---------------------------------------------------------------------------
extern "C" void kernel_launch(void* const* d_in, const int* in_sizes, int n_in,
                              void* d_out, int out_size, void* d_ws, size_t ws_size,
                              hipStream_t stream) {
  (void)in_sizes; (void)n_in; (void)out_size; (void)ws_size;
  const float* x   = (const float*)d_in[0];
  const float* Wa1 = (const float*)d_in[1];
  const float* ba1 = (const float*)d_in[2];
  const float* Wa2 = (const float*)d_in[3];
  const float* ba2 = (const float*)d_in[4];
  const float* Wb1 = (const float*)d_in[5];
  const float* bb1 = (const float*)d_in[6];
  const float* Wb2 = (const float*)d_in[7];
  const float* bb2 = (const float*)d_in[8];
  const float* lng = (const float*)d_in[9];
  const float* lnb = (const float*)d_in[10];
  const float* Wg  = (const float*)d_in[11];
  const float* bg  = (const float*)d_in[12];
  const float* Wz  = (const float*)d_in[13];
  const float* bz  = (const float*)d_in[14];

  char* ws = (char*)d_ws;
  _Float16* WcatT = (_Float16*)(ws + OFF_WCAT);
  float*    bcat  = (float*)   (ws + OFF_BCAT);
  _Float16* aT    = (_Float16*)(ws + OFF_AT);
  _Float16* bT    = (_Float16*)(ws + OFF_BT);
  _Float16* gbuf  = (_Float16*)(ws + OFF_G);
  float*    zbuf  = (float*)   (ws + OFF_Z);
  float*    out   = (float*)d_out;

  pack_weights<<<dim3(192), dim3(128), 0, stream>>>(Wa1, Wa2, Wb1, Wb2, Wg,
                                                    ba1, ba2, bb1, bb2, bg, WcatT, bcat);
  proj_kernel<<<dim3(NN / 16), dim3(128), 0, stream>>>(x, WcatT, bcat, aT, bT, gbuf);
  tri_gemm<<<dim3(4, 4, 16), dim3(256), 0, stream>>>(aT, bT, zbuf);
  ln_out_kernel<<<dim3(NN), dim3(128), 0, stream>>>(zbuf, gbuf, lng, lnb, Wz, bz, out);
}
